// mhsa_77051713290503
// MI455X (gfx1250) — matrix operand order repaired and checked by reading, not compiled, not run on hardware
//
#include <hip/hip_runtime.h>
#include <hip/hip_bf16.h>

#define BB 8
#define NN 1024
#define CC 256
#define HH 8
#define DD 32

typedef __attribute__((ext_vector_type(8)))  __bf16 v8bf;
typedef __attribute__((ext_vector_type(16))) __bf16 v16bf;
typedef __attribute__((ext_vector_type(8)))  float  v8f;
typedef __attribute__((ext_vector_type(4)))  unsigned int u32x4;
typedef __attribute__((ext_vector_type(8)))  int i32x8;
typedef __attribute__((ext_vector_type(4)))  int i32x4;

// Native f32 -> bf16 (hardware RNE convert)
__device__ __forceinline__ __bf16 f2bf(float f) { return (__bf16)f; }

// A-matrix fragment, 16x32 bf16 (MxK), row-major source with leading dim ld.
// Lane l: M = l&15 ; K = {cb..cb+7, cb+16..cb+23}, cb = (l>=16)?8:0  (ISA 7.12.2)
__device__ __forceinline__ v16bf load_fragA(const __bf16* src, int ld, int lane) {
  int m  = lane & 15;
  int cb = (lane >> 4) << 3;
  const __bf16* p = src + m * ld + cb;
  v8bf lo = *(const v8bf*)p;
  v8bf hi = *(const v8bf*)(p + 16);
  v16bf out;
#pragma unroll
  for (int i = 0; i < 8; ++i) { out[i] = lo[i]; out[i + 8] = hi[i]; }
  return out;
}

// B-matrix fragment, 32x16 bf16 (KxN). Source stored as [N][K] row-major
// (i.e. column n of B contiguous in K), leading dim ld.
// Lane l: N = l&15 ; K = {kb..kb+7, kb+16..kb+23}, kb = (l>=16)?8:0 (matches A)
__device__ __forceinline__ v16bf load_fragB(const __bf16* src, int ld, int lane) {
  int n  = lane & 15;
  int kb = (lane >> 4) << 3;
  const __bf16* p = src + n * ld + kb;
  v8bf lo = *(const v8bf*)p;
  v8bf hi = *(const v8bf*)(p + 16);
  v16bf out;
#pragma unroll
  for (int i = 0; i < 8; ++i) { out[i] = lo[i]; out[i + 8] = hi[i]; }
  return out;
}

__device__ __forceinline__ v8f wmma_bf16(v16bf a, v16bf b, v8f c) {
  return __builtin_amdgcn_wmma_f32_16x16x32_bf16(false, a, false, b, (short)0, c,
                                                 false, false);
}

// ---------------- Kernel 1: box decode + position embedding + x = qc + pe ---
__global__ __launch_bounds__(256) void k_pe_x(
    const float* __restrict__ qc, const float* __restrict__ xyzr,
    float* __restrict__ rois, float* __restrict__ xf, __bf16* __restrict__ xb) {
  int row = blockIdx.x;               // b*N + n
  int c = threadIdx.x;                // channel 0..255
  const float* xz = xyzr + (size_t)row * 4;
  float px = xz[0], py = xz[1], pz = xz[2], pr = xz[3];
  if (c == 0) {
    float scale = exp2f(pz);
    float w = scale * exp2f(-0.5f * pr);
    float h = scale * exp2f( 0.5f * pr);
    rois[(size_t)row * 4 + 0] = px - 0.5f * w;
    rois[(size_t)row * 4 + 1] = py - 0.5f * h;
    rois[(size_t)row * 4 + 2] = px + 0.5f * w;
    rois[(size_t)row * 4 + 3] = py + 0.5f * h;
  }
  int i = c >> 6, r = c & 63, k = r >> 1, s = r & 1;
  float tok = (i == 0 ? px : i == 1 ? py : i == 2 ? pz : pr) * (i < 2 ? 0.001f : 1.0f);
  // dim_t = 10000^(k/32) -> inv = 2^(-(k/32)*log2(10000))
  float inv = exp2f(-(float)k * 0.03125f * 13.287712379549449f);
  float p = tok * inv;
  float pe = s ? __cosf(p) : __sinf(p);
  float xv = qc[(size_t)row * CC + c] + pe;
  xf[(size_t)row * CC + c] = xv;
  xb[(size_t)row * CC + c] = f2bf(xv);
}

// ---------------- Kernel 2: log(IoF + 1e-7) bias matrix (L2-resident 32MB) --
__global__ __launch_bounds__(256) void k_iof(const float* __restrict__ rois,
                                             float* __restrict__ iofl) {
  __shared__ float bi[16][4], bj[16][4];
  int b = blockIdx.z;
  int i0 = blockIdx.x * 16, j0 = blockIdx.y * 16;
  int t = threadIdx.x;
  if (t < 16) {
    size_t r = ((size_t)b * NN + i0 + t) * 4;
    bi[t][0] = rois[r]; bi[t][1] = rois[r + 1]; bi[t][2] = rois[r + 2]; bi[t][3] = rois[r + 3];
  } else if (t < 32) {
    int tt = t - 16;
    size_t r = ((size_t)b * NN + j0 + tt) * 4;
    bj[tt][0] = rois[r]; bj[tt][1] = rois[r + 1]; bj[tt][2] = rois[r + 2]; bj[tt][3] = rois[r + 3];
  }
  __syncthreads();
  int ti = t >> 4, tj = t & 15;
  float ax1 = bi[ti][0], ay1 = bi[ti][1], ax2 = bi[ti][2], ay2 = bi[ti][3];
  float area = (ax2 - ax1) * (ay2 - ay1);
  float ltx = fmaxf(ax1, bj[tj][0]), lty = fmaxf(ay1, bj[tj][1]);
  float rbx = fminf(ax2, bj[tj][2]), rby = fminf(ay2, bj[tj][3]);
  float w = fmaxf(rbx - ltx, 0.f), h = fmaxf(rby - lty, 0.f);
  float iof = (w * h) / fmaxf(area, 1e-6f);
  iofl[((size_t)b * NN + i0 + ti) * NN + j0 + tj] = __logf(iof + 1e-7f);
}

// ---------------- Kernel 3: QKV projection GEMM (8192x768x256, bf16 WMMA) ---
// A-tile (128x32 bf16) is fetched by the Tensor Data Mover into padded LDS
// rows (TDM pad: 16 DWORDs data + 4 DWORDs pad == 80B rows == LDA3 elements).
#define LDA3 40
__global__ __launch_bounds__(256) void k_qkv(
    const __bf16* __restrict__ xb, const float* __restrict__ W,
    const float* __restrict__ bias, __bf16* __restrict__ qb,
    __bf16* __restrict__ kb, __bf16* __restrict__ vb) {
  __shared__ __bf16 As[128 * LDA3];
  __shared__ __bf16 Bs[64 * LDA3];
  int m0 = blockIdx.x * 128;
  int n0 = blockIdx.y * 64;                 // 0..767 in steps of 64
  int t = threadIdx.x, w = t >> 5, lane = t & 31;
  v8f acc[4] = {};

  // Loop-invariant D# pieces (ISA cdna5 §8 bit layout)
  unsigned lds_off = (unsigned)(uintptr_t)(void*)As;      // LDS byte offset
  for (int kk = 0; kk < CC; kk += 32) {
    if (w == 0) {   // wave 0 drives the TDM for the A tile
      unsigned long long ga =
          (unsigned long long)(uintptr_t)(xb + (size_t)m0 * CC + kk);
      u32x4 g0;
      g0[0] = 1u;                                   // count=1, user mode
      g0[1] = lds_off;                              // lds_addr
      g0[2] = (unsigned)(ga & 0xffffffffu);         // global_addr[31:0]
      g0[3] = (unsigned)((ga >> 32) & 0x01ffffffu)  // global_addr[56:32]
            | 0x80000000u;                          // type=2 ("image")
      i32x8 g1;
      g1[0] = (1 << 16)        // data_size = 2B
            | (1 << 20)        // pad_enable
            | (3 << 22)        // pad_interval: 16 DWORDs (one 64B row)
            | (3 << 25);       // pad_amount: 4 DWORDs (16B)
      g1[1] = (int)((0x100000u & 0xffffu) << 16);   // tensor_dim0 lo16 (1M)
      g1[2] = (int)((0x100000u >> 16) |             // tensor_dim0 hi16
                    ((0x100000u & 0xffffu) << 16)); // tensor_dim1 lo16
      g1[3] = (int)((0x100000u >> 16) | (32u << 16)); // dim1 hi | tile_dim0=32
      g1[4] = 128;                                  // tile_dim1=128, tile_dim2=0
      g1[5] = CC;                                   // tensor_dim0_stride = 256
      g1[6] = 0;
      g1[7] = 0;
      i32x4 g2; g2[0] = 0x100000; g2[1] = 0; g2[2] = 0; g2[3] = 0;
      i32x4 g3; g3[0] = 0; g3[1] = 0; g3[2] = 0; g3[3] = 0;
      i32x8 g4; g4[0] = 0; g4[1] = 0; g4[2] = 0; g4[3] = 0;
               g4[4] = 0; g4[5] = 0; g4[6] = 0; g4[7] = 0;
      __builtin_amdgcn_tensor_load_to_lds(g0, g1, g2, g3, g4, 0);
      __builtin_amdgcn_s_wait_tensorcnt((short)0);
    }
    {   // stage B tile 64x32: W[kout][c] f32 -> bf16 (all threads)
      int row = t >> 2, q = t & 3;
      const float* g = W + (size_t)(n0 + row) * CC + kk + q * 8;
      __bf16 tmp[8];
#pragma unroll
      for (int i = 0; i < 8; ++i) tmp[i] = f2bf(g[i]);
      *(v8bf*)&Bs[row * LDA3 + q * 8] = *(v8bf*)tmp;
    }
    __syncthreads();
    v16bf a = load_fragA(&As[(w * 16) * LDA3], LDA3, lane);
#pragma unroll
    for (int tn = 0; tn < 4; ++tn) {
      v16bf bf = load_fragB(&Bs[(tn * 16) * LDA3], LDA3, lane);
      acc[tn] = wmma_bf16(a, bf, acc[tn]);
    }
    __syncthreads();
  }
  // epilogue: bias add, route to q/k/v, bf16 store
  int region = n0 >> 8;                     // 0=q 1=k 2=v
  __bf16* outp = region == 0 ? qb : region == 1 ? kb : vb;
  int colbase = n0 & 255;
  int nc = lane & 15, rhi = (lane >> 4) * 8;
#pragma unroll
  for (int tn = 0; tn < 4; ++tn) {
    float bv = bias[n0 + tn * 16 + nc];
    int col = colbase + tn * 16 + nc;
#pragma unroll
    for (int r = 0; r < 8; ++r) {
      int m = m0 + w * 16 + r + rhi;
      outp[(size_t)m * CC + col] = f2bf(acc[tn][r] + bv);
    }
  }
}

// ---------------- Kernel 4: flash attention with IoF bias (per b,h) ---------
__global__ __launch_bounds__(256) void k_attn(
    const __bf16* __restrict__ qb, const __bf16* __restrict__ kb,
    const __bf16* __restrict__ vb, const float* __restrict__ iofl,
    const float* __restrict__ tauv, __bf16* __restrict__ ob) {
  __shared__ __bf16 vt[32 * 40];            // V^T chunk [dim][key]
  __shared__ __bf16 pls[8][16 * 40];        // per-wave P tile [16][32]
  int b = blockIdx.z, h = blockIdx.y;
  int q0 = blockIdx.x * 128;
  int t = threadIdx.x, w = t >> 5, lane = t & 31;
  float tau = tauv[h];
  const float scale = 0.17677669529663687f; // 1/sqrt(32)
  int qrow = q0 + w * 16;
  v16bf qf = load_fragA(qb + ((size_t)b * NN + qrow) * CC + h * DD, CC, lane);
  v8f o0 = {}, o1 = {};
  float mrow[8], lrow[8];
#pragma unroll
  for (int r = 0; r < 8; ++r) { mrow[r] = -1e30f; lrow[r] = 0.f; }
  int nc = lane & 15, rhi = (lane >> 4) * 8;

  for (int key0 = 0; key0 < NN; key0 += 32) {
    __syncthreads();
#pragma unroll
    for (int e = 0; e < 4; ++e) {           // stage V^T (32 keys x 32 dims)
      int lin = t + 256 * e;
      int kky = lin >> 5, dim = lin & 31;
      vt[dim * 40 + kky] = vb[((size_t)b * NN + key0 + kky) * CC + h * DD + dim];
    }
    __syncthreads();
    // S = Q K^T : K rows are already B-fragment columns (contiguous dims)
    v16bf kf0 = load_fragB(kb + ((size_t)b * NN + key0) * CC + h * DD, CC, lane);
    v16bf kf1 = load_fragB(kb + ((size_t)b * NN + key0 + 16) * CC + h * DD, CC, lane);
    v8f s0 = {}, s1 = {};
    s0 = wmma_bf16(qf, kf0, s0);
    s1 = wmma_bf16(qf, kf1, s1);
    // bias + scale (iofl stays in the 192MB L2; prefetch next chunk)
    const float* iofp = iofl + ((size_t)b * NN + qrow + rhi) * NN + key0;
    if (key0 + 32 < NN) __builtin_prefetch(iofp + 32, 0, 1);
    float sv0[8], sv1[8];
#pragma unroll
    for (int r = 0; r < 8; ++r) {
      const float* ip = iofp + (size_t)r * NN;
      sv0[r] = s0[r] * scale + ip[nc] * tau;
      sv1[r] = s1[r] * scale + ip[16 + nc] * tau;
    }
    __bf16* pp = &pls[w][0];
#pragma unroll
    for (int r = 0; r < 8; ++r) {
      float m = fmaxf(sv0[r], sv1[r]);
      for (int off = 8; off; off >>= 1) m = fmaxf(m, __shfl_xor(m, off, 16));
      float mn = fmaxf(mrow[r], m);
      float corr = __expf(mrow[r] - mn);
      float p0 = __expf(sv0[r] - mn), p1 = __expf(sv1[r] - mn);
      float ps = p0 + p1;
      for (int off = 8; off; off >>= 1) ps += __shfl_xor(ps, off, 16);
      lrow[r] = lrow[r] * corr + ps;
      mrow[r] = mn;
      o0[r] *= corr; o1[r] *= corr;
      int prow = r + rhi;
      pp[prow * 40 + nc]      = f2bf(p0);
      pp[prow * 40 + 16 + nc] = f2bf(p1);
    }
    // wave-lockstep: drain LDS stores before cross-lane fragment reload
    asm volatile("s_wait_dscnt 0" ::: "memory");
    v16bf pa  = load_fragA(pp, 40, lane);
    v16bf vf0 = load_fragB(&vt[0], 40, lane);
    v16bf vf1 = load_fragB(&vt[16 * 40], 40, lane);
    o0 = wmma_bf16(pa, vf0, o0);
    o1 = wmma_bf16(pa, vf1, o1);
  }
#pragma unroll
  for (int r = 0; r < 8; ++r) {
    float invl = 1.0f / fmaxf(lrow[r], 1e-30f);
    size_t mg = (size_t)b * NN + qrow + r + rhi;
    ob[mg * CC + h * DD + nc]      = f2bf(o0[r] * invl);
    ob[mg * CC + h * DD + 16 + nc] = f2bf(o1[r] * invl);
  }
}

// ---------------- Kernel 5: out-proj GEMM + residual + LayerNorm ------------
#define LDB5 40
__global__ __launch_bounds__(256) void k_out(
    const __bf16* __restrict__ ob, const float* __restrict__ W,
    const float* __restrict__ bias, const float* __restrict__ xf,
    const float* __restrict__ gamma, const float* __restrict__ beta,
    float* __restrict__ out) {
  __shared__ __bf16 As[32 * LDB5];
  __shared__ __bf16 Bs[256 * LDB5];
  __shared__ float ys[32][264];
  int m0 = blockIdx.x * 32;
  int t = threadIdx.x, w = t >> 5, lane = t & 31;
  int rg = w >> 2, cg = w & 3;
  v8f acc[4] = {};
  for (int kk = 0; kk < CC; kk += 32) {
    if (t < 128) {                          // stage A 32x32
      int row = t >> 2, q = t & 3;
      *(v8bf*)&As[row * LDB5 + q * 8] =
          *(const v8bf*)(ob + (size_t)(m0 + row) * CC + kk + q * 8);
    }
#pragma unroll
    for (int e = 0; e < 4; ++e) {           // stage B 256x32 (f32 -> bf16)
      int idx = t + 256 * e;
      int row = idx >> 2, q = idx & 3;
      const float* g = W + (size_t)row * CC + kk + q * 8;
      __bf16 tmp[8];
#pragma unroll
      for (int i = 0; i < 8; ++i) tmp[i] = f2bf(g[i]);
      *(v8bf*)&Bs[row * LDB5 + q * 8] = *(v8bf*)tmp;
    }
    __syncthreads();
    v16bf a = load_fragA(&As[(rg * 16) * LDB5], LDB5, lane);
#pragma unroll
    for (int tn = 0; tn < 4; ++tn) {
      v16bf bf = load_fragB(&Bs[(cg * 64 + tn * 16) * LDB5], LDB5, lane);
      acc[tn] = wmma_bf16(a, bf, acc[tn]);
    }
    __syncthreads();
  }
  // y = x + (o W^T + b) staged to LDS
  int nc = lane & 15, rhi = (lane >> 4) * 8;
#pragma unroll
  for (int tn = 0; tn < 4; ++tn) {
    int col = cg * 64 + tn * 16 + nc;
    float bv = bias[col];
#pragma unroll
    for (int r = 0; r < 8; ++r) {
      int row = rg * 16 + r + rhi;
      ys[row][col] = acc[tn][r] + bv + xf[(size_t)(m0 + row) * CC + col];
    }
  }
  __syncthreads();
  // LayerNorm: 8 lanes per row, 32 cols each
  int row = t >> 3, sl = t & 7;
  float sum = 0.f, sq = 0.f;
#pragma unroll
  for (int i = 0; i < 32; ++i) {
    float v = ys[row][sl * 32 + i];
    sum += v; sq += v * v;
  }
  for (int off = 4; off; off >>= 1) {
    sum += __shfl_xor(sum, off, 8);
    sq  += __shfl_xor(sq,  off, 8);
  }
  float mean = sum * (1.0f / CC);
  float var  = sq * (1.0f / CC) - mean * mean;
  float rstd = rsqrtf(var + 1e-5f);
#pragma unroll
  for (int i = 0; i < 32; ++i) {
    int c = sl * 32 + i;
    out[(size_t)(m0 + row) * CC + c] =
        (ys[row][c] - mean) * rstd * gamma[c] + beta[c];
  }
}

// ---------------------------------------------------------------------------
extern "C" void kernel_launch(void* const* d_in, const int* in_sizes, int n_in,
                              void* d_out, int out_size, void* d_ws, size_t ws_size,
                              hipStream_t stream) {
  const float* qc    = (const float*)d_in[0];
  const float* xyzr  = (const float*)d_in[1];
  const float* tau   = (const float*)d_in[2];
  const float* w_in  = (const float*)d_in[3];
  const float* b_in  = (const float*)d_in[4];
  const float* w_out = (const float*)d_in[5];
  const float* b_out = (const float*)d_in[6];
  const float* gamma = (const float*)d_in[7];
  const float* beta  = (const float*)d_in[8];
  float* outp = (float*)d_out;

  char* ws = (char*)d_ws;
  size_t off = 0;
  auto take = [&](size_t bytes) { void* p = ws + off; off += (bytes + 255) & ~(size_t)255; return p; };
  float*  rois = (float*) take((size_t)BB * NN * 4 * 4);
  float*  xf   = (float*) take((size_t)BB * NN * CC * 4);
  __bf16* xb   = (__bf16*)take((size_t)BB * NN * CC * 2);
  __bf16* qb   = (__bf16*)take((size_t)BB * NN * CC * 2);
  __bf16* kb   = (__bf16*)take((size_t)BB * NN * CC * 2);
  __bf16* vb   = (__bf16*)take((size_t)BB * NN * CC * 2);
  __bf16* ob   = (__bf16*)take((size_t)BB * NN * CC * 2);
  float*  iofl = (float*) take((size_t)BB * NN * NN * 4);

  k_pe_x<<<dim3(BB * NN), 256, 0, stream>>>(qc, xyzr, rois, xf, xb);
  k_iof <<<dim3(NN / 16, NN / 16, BB), 256, 0, stream>>>(rois, iofl);
  k_qkv <<<dim3(BB * NN / 128, (3 * CC) / 64), 256, 0, stream>>>(xb, w_in, b_in, qb, kb, vb);
  k_attn<<<dim3(NN / 128, HH, BB), 256, 0, stream>>>(qb, kb, vb, iofl, tau, ob);
  k_out <<<dim3(BB * NN / 32), 256, 0, stream>>>(ob, w_out, b_out, xf, gamma, beta, outp);
}
